// GraphSAGE_20607253086683
// MI455X (gfx1250) — matrix operand order repaired and checked by reading, not compiled, not run on hardware
//
#include <hip/hip_runtime.h>
#include <math.h>

// ---------------------------------------------------------------------------
// GraphSAGE (2-layer, weighted-mean aggregate) for MI455X / gfx1250 (wave32).
//  - Edge phase: coalesced float4 gathers + global f32 atomics (memory-bound,
//    accumulators fit in 192MB L2).
//  - Node phase: fused [agg | x] @ [Wl; Wr] via v_wmma_f32_16x16x32_f16.
//    LDS layouts chosen so every WMMA fragment is loaded with ds_load_b128:
//      * A tile row-major [m][k]  -> lane fragment = two contiguous 16B runs
//      * B stored transposed [n][k] -> lane fragment = one contiguous 32B run
// ---------------------------------------------------------------------------

typedef __attribute__((ext_vector_type(16))) _Float16 v16h;
typedef __attribute__((ext_vector_type(8)))  _Float16 v8h;
typedef __attribute__((ext_vector_type(8)))  float    v8f;

#define N_NODES 100000
#define N_EDGES 1600000
#define D_IN    32
#define D_HID   64
#define D_OUT   40
#define D_OUT_P 48   // padded to 3 WMMA col tiles

// ---------------------------------------------------------------- zero fill
__global__ void k_zero(float* __restrict__ p, long long n) {
    long long i = (long long)blockIdx.x * blockDim.x + threadIdx.x;
    long long stride = (long long)gridDim.x * blockDim.x;
    for (; i < n; i += stride) p[i] = 0.0f;
}

// ---------------------------------------------------------------- in-degree
__global__ void k_degree(const int* __restrict__ dst, float* __restrict__ deg) {
    long long e = (long long)blockIdx.x * blockDim.x + threadIdx.x;
    if (e < N_EDGES) atomicAdd(&deg[dst[e]], 1.0f);
}

// ----------------------------------------------- edge gather + scatter-add
// D floats per node; D/4 lanes per edge, each lane moves one float4.
template <int D>
__global__ void k_edge_agg(const float* __restrict__ x,
                           const int*   __restrict__ src,
                           const int*   __restrict__ dst,
                           const float* __restrict__ w,
                           float*       __restrict__ agg) {
    const int lanesPerEdge = D / 4;
    long long idx = (long long)blockIdx.x * blockDim.x + threadIdx.x;
    long long e   = idx / lanesPerEdge;
    int       c4  = (int)(idx % lanesPerEdge) * 4;
    if (e >= N_EDGES) return;
    int   s  = src[e];
    int   d  = dst[e];
    float we = w[e];
    const float4 xv = *(const float4*)(x + (long long)s * D + c4);
    float* ap = agg + (long long)d * D + c4;
    atomicAdd(ap + 0, xv.x * we);
    atomicAdd(ap + 1, xv.y * we);
    atomicAdd(ap + 2, xv.z * we);
    atomicAdd(ap + 3, xv.w * we);
}

// deterministic per-element hash for the dropout mask
__device__ __forceinline__ unsigned hash_u32(unsigned h) {
    h ^= h >> 16; h *= 0x7feb352du;
    h ^= h >> 15; h *= 0x846ca68bu;
    h ^= h >> 16;
    return h;
}

// load a 16-bit A-matrix fragment from a row-major [m][k] tile:
// lane L (half = L>>4, m = L&15) needs K = {kb+16h..+15} (contiguous, matches B)
// on row m -> two aligned 16-byte LDS loads.
__device__ __forceinline__ v16h load_a_frag(const _Float16* rowPtr, int kb, int half) {
    union { v16h v; v8h h[2]; } a;
    a.h[0] = *(const v8h*)(rowPtr + kb + 16 * half);
    a.h[1] = *(const v8h*)(rowPtr + kb + 16 * half + 8);
    return a.v;
}

// ------------------------------------------------------- layer 1 node phase
// H = dropout(relu(l2norm([agg/deg | X] @ [Wl1;Wr1] + b1)))
// 2 waves / block, one 16-row tile per wave.  K=64 (2 wmma steps), 4 col tiles.
__global__ void __launch_bounds__(64)
k_layer1(const float* __restrict__ X,  const float* __restrict__ agg,
         const float* __restrict__ deg,
         const float* __restrict__ Wl, const float* __restrict__ Wr,
         const float* __restrict__ bias, float* __restrict__ H) {
    __shared__ alignas(32) _Float16 sBt[64][64];     // TRANSPOSED weights [n][k]
    __shared__ alignas(32) _Float16 sA[2][16][64];   // per-wave A tile [m][k]
    __shared__ alignas(32) float    sO[2][16][64];   // per-wave output tile
    __shared__ float    sScale[2][16];

    const int tid  = threadIdx.x;
    const int wave = tid >> 5;
    const int lane = tid & 31;
    const int tile = blockIdx.x * 2 + wave;  // 3125 blocks * 2 = 6250 tiles exact
    const int rowBase = tile * 16;

    // weights, transposed into [n][k]: k rows 0..31 = Wl1, 32..63 = Wr1
    for (int i = tid; i < 64 * 64; i += 64) {
        int k = i >> 6, n = i & 63;   // n fast -> coalesced global reads
        float v = (k < 32) ? Wl[k * 64 + n] : Wr[(k - 32) * 64 + n];
        sBt[n][k] = (_Float16)v;
    }

    // build A tile: cols 0..31 = agg/deg, cols 32..63 = X
    for (int r = 0; r < 16; ++r) {
        int row = rowBase + r;
        float dg = fmaxf(deg[row], 1.0f);
        sA[wave][r][lane]      = (_Float16)(agg[(long long)row * 32 + lane] / dg);
        sA[wave][r][lane + 32] = (_Float16)(X[(long long)row * 32 + lane]);
    }
    __syncthreads();

    const int half = lane >> 4;
    const int m    = lane & 15;
    const _Float16* aRow = &sA[wave][m][0];

    v16h afr[2];
    afr[0] = load_a_frag(aRow, 0, half);
    afr[1] = load_a_frag(aRow, 32, half);

    #pragma unroll
    for (int t = 0; t < 4; ++t) {            // 4 column tiles of 16
        const _Float16* bRow = &sBt[t * 16 + m][0];
        v8f c = {};
        #pragma unroll
        for (int kc = 0; kc < 2; ++kc) {
            // B fragment: 16 consecutive K along transposed row -> one 32B run
            v16h bfr = *(const v16h*)(bRow + kc * 32 + 16 * half);
            c = __builtin_amdgcn_wmma_f32_16x16x32_f16(
                    false, afr[kc], false, bfr, (short)0, c, false, false);
        }
        float bv = bias[t * 16 + m];
        #pragma unroll
        for (int j = 0; j < 8; ++j)          // D: VGPR j -> row j + 8*half, col m
            sO[wave][j + 8 * half][t * 16 + m] = c[j] + bv;
    }
    __syncthreads();

    // row-wise L2 normalize
    if (lane < 16) {
        float ss = 0.0f;
        for (int n = 0; n < 64; ++n) { float v = sO[wave][lane][n]; ss += v * v; }
        sScale[wave][lane] = rsqrtf(fmaxf(ss, 1e-24f));
    }
    __syncthreads();

    // relu + deterministic dropout (p=0.5 -> scale 2x) + store
    for (int r = 0; r < 16; ++r) {
        int row = rowBase + r;
        float sc = sScale[wave][r];
        #pragma unroll
        for (int p = 0; p < 2; ++p) {
            int cc = lane + p * 32;
            float v = fmaxf(sO[wave][r][cc] * sc, 0.0f);
            unsigned h = hash_u32((unsigned)(row * 64 + cc) ^ 0x9E3779B9u);
            v = (h & 1u) ? v * 2.0f : 0.0f;
            H[(long long)row * 64 + cc] = v;
        }
    }
}

// ------------------------------------------------------- layer 2 node phase
// Z = [agg2/deg | H] @ [Wl2;Wr2] + b2 ; out = log_softmax(Z)
// K=128 (4 wmma steps), 3 col tiles (40 cols padded to 48).
__global__ void __launch_bounds__(64)
k_layer2(const float* __restrict__ H,  const float* __restrict__ agg,
         const float* __restrict__ deg,
         const float* __restrict__ Wl, const float* __restrict__ Wr,
         const float* __restrict__ bias, float* __restrict__ out) {
    __shared__ alignas(32) _Float16 sBt[D_OUT_P][128];  // TRANSPOSED [n][k]
    __shared__ alignas(32) _Float16 sA[2][16][128];
    __shared__ alignas(32) float    sO[2][16][D_OUT_P];
    __shared__ float    sRed[2][16][2];      // [0]=rowmax  [1]=log(sumexp)

    const int tid  = threadIdx.x;
    const int wave = tid >> 5;
    const int lane = tid & 31;
    const int tile = blockIdx.x * 2 + wave;
    const int rowBase = tile * 16;

    // weights transposed: k rows 0..63 = Wl2 (64x40), 64..127 = Wr2; n>=40 zero
    for (int i = tid; i < 128 * D_OUT_P; i += 64) {
        int k = i / D_OUT_P, n = i % D_OUT_P;
        float v = 0.0f;
        if (n < D_OUT) v = (k < 64) ? Wl[k * D_OUT + n] : Wr[(k - 64) * D_OUT + n];
        sBt[n][k] = (_Float16)v;
    }

    // A tile: cols 0..63 = agg2/deg, cols 64..127 = H
    for (int r = 0; r < 16; ++r) {
        int row = rowBase + r;
        float dg = fmaxf(deg[row], 1.0f);
        #pragma unroll
        for (int p = 0; p < 2; ++p) {
            int cc = lane + p * 32;
            sA[wave][r][cc]      = (_Float16)(agg[(long long)row * 64 + cc] / dg);
            sA[wave][r][cc + 64] = (_Float16)(H[(long long)row * 64 + cc]);
        }
    }
    __syncthreads();

    const int half = lane >> 4;
    const int m    = lane & 15;
    const _Float16* aRow = &sA[wave][m][0];

    v16h afr[4];
    #pragma unroll
    for (int kc = 0; kc < 4; ++kc)
        afr[kc] = load_a_frag(aRow, kc * 32, half);

    #pragma unroll
    for (int t = 0; t < 3; ++t) {
        const _Float16* bRow = &sBt[t * 16 + m][0];
        v8f c = {};
        #pragma unroll
        for (int kc = 0; kc < 4; ++kc) {
            v16h bfr = *(const v16h*)(bRow + kc * 32 + 16 * half);
            c = __builtin_amdgcn_wmma_f32_16x16x32_f16(
                    false, afr[kc], false, bfr, (short)0, c, false, false);
        }
        int n = t * 16 + m;
        float bv = (n < D_OUT) ? bias[n] : 0.0f;
        #pragma unroll
        for (int j = 0; j < 8; ++j)
            sO[wave][j + 8 * half][n] = c[j] + bv;
    }
    __syncthreads();

    // log_softmax over the 40 valid columns
    if (lane < 16) {
        float mx = -3.0e38f;
        for (int n = 0; n < D_OUT; ++n) mx = fmaxf(mx, sO[wave][lane][n]);
        float s = 0.0f;
        for (int n = 0; n < D_OUT; ++n) s += __expf(sO[wave][lane][n] - mx);
        sRed[wave][lane][0] = mx;
        sRed[wave][lane][1] = __logf(s);
    }
    __syncthreads();

    for (int r = 0; r < 16; ++r) {
        int row = rowBase + r;
        float mx = sRed[wave][r][0];
        float ls = sRed[wave][r][1];
        out[(long long)row * D_OUT + lane] = sO[wave][r][lane] - mx - ls;
        if (lane < D_OUT - 32)
            out[(long long)row * D_OUT + lane + 32] = sO[wave][r][lane + 32] - mx - ls;
    }
}

// ---------------------------------------------------------------------------
extern "C" void kernel_launch(void* const* d_in, const int* in_sizes, int n_in,
                              void* d_out, int out_size, void* d_ws, size_t ws_size,
                              hipStream_t stream) {
    const float* X   = (const float*)d_in[0];
    const int*   A   = (const int*)  d_in[1];   // [2, E]: row0=src, row1=dst
    const float* W   = (const float*)d_in[2];
    const float* Wl1 = (const float*)d_in[3];
    const float* Wr1 = (const float*)d_in[4];
    const float* b1  = (const float*)d_in[5];
    const float* Wl2 = (const float*)d_in[6];
    const float* Wr2 = (const float*)d_in[7];
    const float* b2  = (const float*)d_in[8];
    float* out = (float*)d_out;

    const int* src = A;
    const int* dst = A + N_EDGES;

    // workspace: deg[N] | agg1[N*32] | agg2[N*64] | H[N*64]
    float* ws   = (float*)d_ws;
    float* deg  = ws;
    float* agg1 = deg  + N_NODES;
    float* agg2 = agg1 + (size_t)N_NODES * 32;
    float* H    = agg2 + (size_t)N_NODES * 64;

    // zero the accumulators (deg|agg1|agg2 are contiguous: N*97 floats)
    k_zero<<<2048, 256, 0, stream>>>(ws, (long long)N_NODES * 97);

    k_degree<<<(N_EDGES + 255) / 256, 256, 0, stream>>>(dst, deg);

    {   // layer-1 edge phase: E * (32/4) lanes
        long long T = (long long)N_EDGES * 8;
        k_edge_agg<32><<<(unsigned)((T + 255) / 256), 256, 0, stream>>>(X, src, dst, W, agg1);
    }
    k_layer1<<<N_NODES / 32, 64, 0, stream>>>(X, agg1, deg, Wl1, Wr1, b1, H);

    {   // layer-2 edge phase: E * (64/4) lanes
        long long T = (long long)N_EDGES * 16;
        k_edge_agg<64><<<(unsigned)((T + 255) / 256), 256, 0, stream>>>(H, src, dst, W, agg2);
    }
    k_layer2<<<N_NODES / 32, 64, 0, stream>>>(H, agg2, deg, Wl2, Wr2, b2, out);
}
